// BiGraphContrastLayer_2911987826804
// MI455X (gfx1250) — matrix operand order repaired and checked by reading, not compiled, not run on hardware
//
#include <hip/hip_runtime.h>
#include <hip/hip_bf16.h>

#define NS_   10000
#define ND_   10000
#define NTOT  (NS_ + ND_)
#define EDGES 320000
#define DIN   512
#define NH    8
#define DHD   64
#define FEAT  (NH * DHD)   // 512
#define SLOPE 0.2f

typedef __attribute__((ext_vector_type(16))) __bf16 v16bf;
typedef __attribute__((ext_vector_type(8)))  float  v8f;

static __device__ __forceinline__ unsigned short f2bf(float f) {
    unsigned u = __float_as_uint(f);
    u += 0x7fffu + ((u >> 16) & 1u);      // round to nearest even
    return (unsigned short)(u >> 16);
}

static __device__ __forceinline__ float leaky(float v) {
    return v > 0.f ? v : SLOPE * v;
}

// ---------------- conversion kernels ----------------

__global__ void conv_x_kernel(const float* __restrict__ xs, const float* __restrict__ xd,
                              unsigned short* __restrict__ xb) {
    int i = blockIdx.x * blockDim.x + threadIdx.x;
    if (i >= NTOT * DIN) return;
    int n = i >> 9;   // / DIN
    float v = (n < NS_) ? xs[i] : xd[i - NS_ * DIN];
    xb[i] = f2bf(v);
}

// W is [K=512][N=512] row-major; store wt as [N][K] (column-major W) in bf16
__global__ void conv_w_kernel(const float* __restrict__ W, unsigned short* __restrict__ wt) {
    int i = blockIdx.x * blockDim.x + threadIdx.x;
    if (i >= DIN * FEAT) return;
    int n = i & (FEAT - 1);
    int k = i >> 9;
    wt[n * DIN + k] = f2bf(W[k * FEAT + n]);
}

// ---------------- WMMA GEMM: z[NTOT,512] = xb @ W (bf16 in, f32 out) ----------------
// Register-blocked: each wave computes a 32x64 block = 2 M-tiles x 4 N-tiles.
// Workgroup = 8 waves = one 32-row band x all 512 columns. 625 workgroups.

__global__ __launch_bounds__(256) void gemm_bf16_wmma(const unsigned short* __restrict__ xb,
                                                      const unsigned short* __restrict__ wt,
                                                      float* __restrict__ z) {
    int bm   = blockIdx.x;            // 0..624  (32-row band)
    int bn   = threadIdx.x >> 5;      // 0..7    (64-col slab per wave)
    int lane = threadIdx.x & 31;
    int hi   = lane >> 4;             // half-wave select
    int l15  = lane & 15;

    union AB { v16bf v; uint4 u[2]; };

    const unsigned short* ap[2];
    const unsigned short* bp[4];
#pragma unroll
    for (int i = 0; i < 2; ++i)
        ap[i] = xb + (size_t)(bm * 32 + i * 16 + l15) * DIN;
#pragma unroll
    for (int j = 0; j < 4; ++j)
        bp[j] = wt + (size_t)(bn * 64 + j * 16 + l15) * DIN;

    v8f c[2][4];
#pragma unroll
    for (int i = 0; i < 2; ++i)
#pragma unroll
        for (int j = 0; j < 4; ++j)
            c[i][j] = (v8f){0.f, 0.f, 0.f, 0.f, 0.f, 0.f, 0.f, 0.f};

    for (int k = 0; k < DIN; k += 32) {
        int ka = k + hi * 8;    // A: lanes<16 -> K k..k+7 & k+16..k+23 ; lanes>=16 -> +8
        int kb = k + hi * 8;    // B: lanes<16 -> K k..k+7 & k+16..k+23 ; lanes>=16 -> +8
        AB a[2], b[4];
#pragma unroll
        for (int i = 0; i < 2; ++i) {
            a[i].u[0] = *(const uint4*)(ap[i] + ka);
            a[i].u[1] = *(const uint4*)(ap[i] + ka + 16);
        }
#pragma unroll
        for (int j = 0; j < 4; ++j) {
            b[j].u[0] = *(const uint4*)(bp[j] + kb);
            b[j].u[1] = *(const uint4*)(bp[j] + kb + 16);
        }
#pragma unroll
        for (int i = 0; i < 2; ++i)
#pragma unroll
            for (int j = 0; j < 4; ++j)
                c[i][j] = __builtin_amdgcn_wmma_f32_16x16x32_bf16(
                    false, a[i].v, false, b[j].v, (short)0, c[i][j], false, false);
    }

    // C layout: VGPR r -> (M = r + 8*hi, N = lane&15)
#pragma unroll
    for (int i = 0; i < 2; ++i) {
        int m0 = bm * 32 + i * 16 + hi * 8;
#pragma unroll
        for (int j = 0; j < 4; ++j) {
            int n = bn * 64 + j * 16 + l15;
#pragma unroll
            for (int r = 0; r < 8; ++r)
                z[(size_t)(m0 + r) * FEAT + n] = c[i][j][r];
        }
    }
}

// ---------------- el / er : per-(node, head) dot with attn vectors ----------------

__global__ void attn_dot_kernel(const float* __restrict__ z,
                                const float* __restrict__ attn_l,
                                const float* __restrict__ attn_r,
                                float* __restrict__ el, float* __restrict__ er) {
    int idx = blockIdx.x * blockDim.x + threadIdx.x;
    if (idx >= NTOT * NH) return;
    int n = idx >> 3;
    int h = idx & 7;
    const float4* zp = (const float4*)(z + (size_t)n * FEAT + h * DHD);
    const float4* al = (const float4*)(attn_l + h * DHD);
    const float4* ar = (const float4*)(attn_r + h * DHD);
    float sl = 0.f, sr = 0.f;
#pragma unroll
    for (int i = 0; i < DHD / 4; ++i) {
        float4 zv = zp[i], av = al[i], rv = ar[i];
        sl += zv.x * av.x + zv.y * av.y + zv.z * av.z + zv.w * av.w;
        sr += zv.x * rv.x + zv.y * rv.y + zv.z * rv.z + zv.w * rv.w;
    }
    el[idx] = sl;
    er[idx] = sr;
}

// ---------------- CSR build ----------------

__global__ void zero_ints_kernel(int* __restrict__ a, int* __restrict__ b, int n) {
    int i = blockIdx.x * blockDim.x + threadIdx.x;
    if (i < n) { a[i] = 0; b[i] = 0; }
}

__global__ void count_deg_kernel(const int* __restrict__ edge_dst, int* __restrict__ deg) {
    int e = blockIdx.x * blockDim.x + threadIdx.x;
    if (e < EDGES) atomicAdd(&deg[edge_dst[e]], 1);
}

__global__ void scan_excl_kernel(const int* __restrict__ deg, int* __restrict__ off, int n) {
    __shared__ int buf[1024];
    __shared__ int carry;
    int t = threadIdx.x;
    if (t == 0) carry = 0;
    __syncthreads();
    for (int base = 0; base < n; base += 1024) {
        int i = base + t;
        int v = (i < n) ? deg[i] : 0;
        buf[t] = v;
        __syncthreads();
        for (int o = 1; o < 1024; o <<= 1) {
            int tmp = (t >= o) ? buf[t - o] : 0;
            __syncthreads();
            buf[t] += tmp;
            __syncthreads();
        }
        int c = carry;
        if (i < n) off[i] = c + buf[t] - v;   // exclusive
        __syncthreads();
        if (t == 1023) carry = c + buf[1023];
        __syncthreads();
    }
    if (t == 0) off[n] = carry;
}

__global__ void scatter_kernel(const int* __restrict__ edge_src, const int* __restrict__ edge_dst,
                               const int* __restrict__ off, int* __restrict__ cursor,
                               int* __restrict__ csr) {
    int e = blockIdx.x * blockDim.x + threadIdx.x;
    if (e >= EDGES) return;
    int d = edge_dst[e];
    int p = atomicAdd(&cursor[d], 1);
    csr[off[d] + p] = edge_src[e];
}

// ---------------- per-dst-node softmax + weighted aggregation ----------------

__global__ __launch_bounds__(256) void gat_aggregate(const float* __restrict__ z,
                                                     const float* __restrict__ el,
                                                     const float* __restrict__ er,
                                                     const int* __restrict__ offp,
                                                     const int* __restrict__ csr,
                                                     const float* __restrict__ bias,
                                                     float* __restrict__ out) {
    __shared__ float wbuf[256 * NH];   // per-edge weights / reduction scratch
    __shared__ int   sidx[256];
    __shared__ float m_sh[NH], s_sh[NH];

    int d = blockIdx.x;
    int t = threadIdx.x;
    int node = NS_ + d;

    float erd[NH], eself[NH];
#pragma unroll
    for (int h = 0; h < NH; ++h) {
        erd[h] = er[node * NH + h];
        eself[h] = leaky(el[node * NH + h] + erd[h]);
    }
    int beg = offp[d];
    int deg = offp[d + 1] - beg;

    // ---- pass 1: per-head max (self-loop included via init) ----
    float p[NH];
#pragma unroll
    for (int h = 0; h < NH; ++h) p[h] = eself[h];
    for (int i = t; i < deg; i += 256) {
        int s = csr[beg + i];
#pragma unroll
        for (int h = 0; h < NH; ++h)
            p[h] = fmaxf(p[h], leaky(el[s * NH + h] + erd[h]));
    }
#pragma unroll
    for (int h = 0; h < NH; ++h) wbuf[t * NH + h] = p[h];
    __syncthreads();
    for (int o = 128; o > 0; o >>= 1) {
        if (t < o)
#pragma unroll
            for (int h = 0; h < NH; ++h)
                wbuf[t * NH + h] = fmaxf(wbuf[t * NH + h], wbuf[(t + o) * NH + h]);
        __syncthreads();
    }
    if (t == 0)
#pragma unroll
        for (int h = 0; h < NH; ++h) m_sh[h] = wbuf[h];
    __syncthreads();
    float m[NH];
#pragma unroll
    for (int h = 0; h < NH; ++h) m[h] = m_sh[h];

    // ---- pass 2: per-head sum of exp ----
#pragma unroll
    for (int h = 0; h < NH; ++h) p[h] = 0.f;
    for (int i = t; i < deg; i += 256) {
        int s = csr[beg + i];
#pragma unroll
        for (int h = 0; h < NH; ++h)
            p[h] += __expf(leaky(el[s * NH + h] + erd[h]) - m[h]);
    }
#pragma unroll
    for (int h = 0; h < NH; ++h) wbuf[t * NH + h] = p[h];
    __syncthreads();
    for (int o = 128; o > 0; o >>= 1) {
        if (t < o)
#pragma unroll
            for (int h = 0; h < NH; ++h)
                wbuf[t * NH + h] += wbuf[(t + o) * NH + h];
        __syncthreads();
    }
    if (t == 0)
#pragma unroll
        for (int h = 0; h < NH; ++h) s_sh[h] = wbuf[h] + __expf(eself[h] - m_sh[h]);
    __syncthreads();

    // ---- pass 3: weighted gather; each thread owns features t and t+256 ----
    int f1 = t, f2 = t + 256;
    int h1 = f1 >> 6, h2 = f2 >> 6;
    float acc1 = __expf(eself[h1] - m[h1]) * z[(size_t)node * FEAT + f1];
    float acc2 = __expf(eself[h2] - m[h2]) * z[(size_t)node * FEAT + f2];

    for (int base = 0; base < deg; base += 256) {
        int cnt = deg - base;
        if (cnt > 256) cnt = 256;
        __syncthreads();
        if (t < cnt) {
            int s = csr[beg + base + t];
            sidx[t] = s;
#pragma unroll
            for (int h = 0; h < NH; ++h)
                wbuf[t * NH + h] = __expf(leaky(el[s * NH + h] + erd[h]) - m[h]);
        }
        __syncthreads();
        for (int j = 0; j < cnt; ++j) {
            int s = sidx[j];
            acc1 += wbuf[j * NH + h1] * z[(size_t)s * FEAT + f1];
            acc2 += wbuf[j * NH + h2] * z[(size_t)s * FEAT + f2];
        }
    }

    out[(size_t)d * FEAT + f1] = acc1 / s_sh[h1] + bias[f1];
    out[(size_t)d * FEAT + f2] = acc2 / s_sh[h2] + bias[f2];
}

// ---------------- launcher ----------------

extern "C" void kernel_launch(void* const* d_in, const int* in_sizes, int n_in,
                              void* d_out, int out_size, void* d_ws, size_t ws_size,
                              hipStream_t stream) {
    (void)in_sizes; (void)n_in; (void)out_size; (void)ws_size;
    const float* x_src    = (const float*)d_in[0];
    const float* x_dst    = (const float*)d_in[1];
    const int*   edge_src = (const int*)d_in[2];
    const int*   edge_dst = (const int*)d_in[3];
    const float* W        = (const float*)d_in[4];
    const float* attn_l   = (const float*)d_in[5];
    const float* attn_r   = (const float*)d_in[6];
    const float* bias     = (const float*)d_in[7];
    float* out = (float*)d_out;

    char* ws = (char*)d_ws;
    size_t pos = 0;
    auto take = [&](size_t bytes) -> char* {
        char* p = ws + pos;
        pos = (pos + bytes + 255) & ~(size_t)255;
        return p;
    };
    unsigned short* xb   = (unsigned short*)take((size_t)NTOT * DIN * 2);
    unsigned short* wt   = (unsigned short*)take((size_t)DIN * FEAT * 2);
    float* z             = (float*)take((size_t)NTOT * FEAT * 4);
    float* el            = (float*)take((size_t)NTOT * NH * 4);
    float* er            = (float*)take((size_t)NTOT * NH * 4);
    int*   deg           = (int*)take((size_t)ND_ * 4);
    int*   cursor        = (int*)take((size_t)ND_ * 4);
    int*   offp          = (int*)take((size_t)(ND_ + 1) * 4);
    int*   csr           = (int*)take((size_t)EDGES * 4);

    conv_x_kernel<<<(NTOT * DIN + 255) / 256, 256, 0, stream>>>(x_src, x_dst, xb);
    conv_w_kernel<<<(DIN * FEAT + 255) / 256, 256, 0, stream>>>(W, wt);

    // 625 workgroups x 8 waves; each wave computes a 32x64 output block
    gemm_bf16_wmma<<<NTOT / 32, 256, 0, stream>>>(xb, wt, z);

    attn_dot_kernel<<<(NTOT * NH + 255) / 256, 256, 0, stream>>>(z, attn_l, attn_r, el, er);

    zero_ints_kernel<<<(ND_ + 255) / 256, 256, 0, stream>>>(deg, cursor, ND_);
    count_deg_kernel<<<(EDGES + 255) / 256, 256, 0, stream>>>(edge_dst, deg);
    scan_excl_kernel<<<1, 1024, 0, stream>>>(deg, offp, ND_);
    scatter_kernel<<<(EDGES + 255) / 256, 256, 0, stream>>>(edge_src, edge_dst, offp, cursor, csr);

    gat_aggregate<<<ND_, 256, 0, stream>>>(z, el, er, offp, csr, bias, out);
}
